// QuantumVariationalClassifier_19224273617445
// MI455X (gfx1250) — matrix operand order repaired and checked by reading, not compiled, not run on hardware
//
#include <hip/hip_runtime.h>
#include <hip/hip_bf16.h>

#define NQUBIT 14
#define NLAYER 4
#define NCLASS 10
#define BATCH 64
#define DIM (1 << NQUBIT)                              // 16384
#define NPARAMS (NLAYER * 3 * NQUBIT + 2 * NQUBIT)     // 196
#define SIM_THREADS 1024                               // 32 waves/WG: min sweep latency

struct c32 { float x, y; };
__device__ inline c32 cmul(c32 a, c32 b) { return { a.x*b.x - a.y*b.y, a.x*b.y + a.y*b.x }; }
__device__ inline c32 cadd(c32 a, c32 b) { return { a.x + b.x, a.y + b.y }; }

// Apply a fused 2x2 complex gate at bit position bp to the LDS-resident state.
__device__ inline void apply_gate(float2* s, int bp,
                                  c32 u00, c32 u01, c32 u10, c32 u11,
                                  int tid, int nthr) {
    const unsigned mask = 1u << bp;
    for (int t = tid; t < (DIM / 2); t += nthr) {
        unsigned lo = (unsigned)t & (mask - 1u);
        unsigned i0 = (((unsigned)t >> bp) << (bp + 1)) | lo;
        unsigned i1 = i0 | mask;
        float2 a = s[i0], b = s[i1];                    // ds_load_b64
        c32 z0 = { a.x, a.y }, z1 = { b.x, b.y };
        c32 n0 = cadd(cmul(u00, z0), cmul(u01, z1));
        c32 n1 = cadd(cmul(u10, z0), cmul(u11, z1));
        s[i0] = make_float2(n0.x, n0.y);                // ds_store_b64
        s[i1] = make_float2(n1.x, n1.y);
    }
    __syncthreads();
}

// One workgroup per batch element; TWO full 2^14 complex state buffers live in
// LDS (2 x 128KB, legal only on CDNA5's 320KB/WGP LDS). The second buffer lets
// us apply the whole 13-CNOT chain as ONE gather-permutation sweep.
__global__ void qvc_sim(const float* __restrict__ x, float* __restrict__ qout) {
    extern __shared__ float smem[];
    float2* buf0 = (float2*)smem;                 // DIM float2 = 128KB
    float2* buf1 = (float2*)(smem + 2 * DIM);     // DIM float2 = 128KB
    float*  xs   = smem + 4 * DIM;                // 196 params
    float*  red  = xs + 200;                      // 16-float reduction buffer

    const int b    = blockIdx.x;
    const int tid  = threadIdx.x;
    const int nthr = blockDim.x;

    __builtin_prefetch(&x[b * NPARAMS], 0, 3);    // global_prefetch_b8

    for (int i = tid; i < NPARAMS; i += nthr) xs[i] = x[b * NPARAMS + i];
    if (tid < 16) red[tid] = 0.0f;
    for (int i = tid; i < DIM; i += nthr) buf0[i] = make_float2(0.0f, 0.0f);
    __syncthreads();
    if (tid == 0) buf0[0] = make_float2(1.0f, 0.0f);
    __syncthreads();

    float2* cur = buf0;
    float2* alt = buf1;

    int p = 0;
    for (int layer = 0; layer < NLAYER; ++layer) {
        for (int q = 0; q < NQUBIT; ++q) {
            float ta = xs[p], tb = xs[p + 1], tc = xs[p + 2]; p += 3;
            float ca, sa, cb, sb, cc, sc;
            __sincosf(0.5f * ta, &sa, &ca);
            __sincosf(0.5f * tb, &sb, &cb);
            __sincosf(0.5f * tc, &sc, &cc);
            // M = Ry(tb) * Rx(ta)
            c32 m00 = {  cb * ca,  sb * sa };
            c32 m01 = { -sb * ca, -cb * sa };
            c32 m10 = {  sb * ca, -cb * sa };
            c32 m11 = {  cb * ca, -sb * sa };
            // U = Rz(tc) * M ; row0 *= e^{-i tc/2}, row1 *= e^{+i tc/2}
            c32 em = { cc, -sc }, ep = { cc, sc };
            apply_gate(cur, NQUBIT - 1 - q,
                       cmul(em, m00), cmul(em, m01),
                       cmul(ep, m10), cmul(ep, m11), tid, nthr);
        }
        // Entire adjacent-qubit CNOT chain as ONE permutation gather:
        //   final[i] = cur[(g_0 o g_1 o ... o g_{12})(i)]
        //   g_q(j) = j ^ (bit_{bt+1}(j) << bt), bt = NQUBIT-2-q, applied bt=0..12.
        for (int i = tid; i < DIM; i += nthr) {
            unsigned j = (unsigned)i;
#pragma unroll
            for (int bt = 0; bt <= NQUBIT - 2; ++bt)
                j ^= ((j >> (bt + 1)) & 1u) << bt;
            alt[i] = cur[j];
        }
        __syncthreads();
        float2* t2 = cur; cur = alt; alt = t2;
    }
    // Final partial layer: Rx then Ry (no Rz)
    for (int q = 0; q < NQUBIT; ++q) {
        float ta = xs[p], tb = xs[p + 1]; p += 2;
        float ca, sa, cb, sb;
        __sincosf(0.5f * ta, &sa, &ca);
        __sincosf(0.5f * tb, &sb, &cb);
        c32 m00 = {  cb * ca,  sb * sa };
        c32 m01 = { -sb * ca, -cb * sa };
        c32 m10 = {  sb * ca, -cb * sa };
        c32 m11 = {  cb * ca, -sb * sa };
        apply_gate(cur, NQUBIT - 1 - q, m00, m01, m10, m11, tid, nthr);
    }

    // Pauli-Z expectations: exp_q = sum_i |s_i|^2 * (1 - 2*bit_{13-q}(i))
    float acc[NQUBIT];
#pragma unroll
    for (int q = 0; q < NQUBIT; ++q) acc[q] = 0.0f;
    for (int i = tid; i < DIM; i += nthr) {
        float2 v = cur[i];
        float pr = v.x * v.x + v.y * v.y;
#pragma unroll
        for (int q = 0; q < NQUBIT; ++q)
            acc[q] += ((i >> (NQUBIT - 1 - q)) & 1) ? -pr : pr;
    }
    // wave32 butterfly reduction, then LDS float atomics (ds_add_f32)
#pragma unroll
    for (int q = 0; q < NQUBIT; ++q) {
#pragma unroll
        for (int off = 16; off > 0; off >>= 1)
            acc[q] += __shfl_xor(acc[q], off, 32);
    }
    if ((tid & 31) == 0) {
#pragma unroll
        for (int q = 0; q < NQUBIT; ++q) atomicAdd(&red[q], acc[q]);
    }
    __syncthreads();
    if (tid < 16) qout[b * 16 + tid] = (tid < NQUBIT) ? red[tid] : 0.0f;  // K-padded row
}

// ---- WMMA head: out[64x10] = qout[64x14(pad16)] @ W^T[14x10] + b ----
typedef __attribute__((ext_vector_type(16))) _Float16 v16h;
typedef __attribute__((ext_vector_type(8)))  float    v8f;

__global__ void qvc_linear(const float* __restrict__ qout, const float* __restrict__ W,
                           const float* __restrict__ bias, float* __restrict__ out) {
    const int lane = threadIdx.x & 31;
    const int wave = threadIdx.x >> 5;          // 0..3 -> 16-row tile of the batch

    // A (16x32 f16): lane<16 holds row M=lane, K = {0..7, 16..23};
    //                lane>=16 holds row M=lane-16, K = {8..15, 24..31}  (ISA 7.12.2)
    const int mrow  = 16 * wave + (lane & 15);
    const int kbase = (lane < 16) ? 0 : 8;
    v16h A, B;
    v8f  C = {};
#pragma unroll
    for (int h = 0; h < 16; ++h) {
        int k = kbase + h + ((h >= 8) ? 8 : 0);
        float av = (k < NQUBIT) ? qout[mrow * 16 + k] : 0.0f;
        A[h] = (_Float16)av;
    }
    // B (32x16 f16): lane<16 holds col N=lane, K = {0..7, 16..23}; lane>=16 col N=lane-16, K = {8..15, 24..31}
    const int ncol = lane & 15;
    const int kb2  = (lane < 16) ? 0 : 8;
#pragma unroll
    for (int h = 0; h < 16; ++h) {
        int k = kb2 + h + ((h >= 8) ? 8 : 0);
        float bv = (k < NQUBIT && ncol < NCLASS) ? W[ncol * NQUBIT + k] : 0.0f;
        B[h] = (_Float16)bv;
    }
    C = __builtin_amdgcn_wmma_f32_16x16x32_f16(false, A, false, B, (short)0, C, false, false);

    float bval = (ncol < NCLASS) ? bias[ncol] : 0.0f;
#pragma unroll
    for (int v = 0; v < 8; ++v) {
        int m   = v + ((lane < 16) ? 0 : 8);    // C/D layout: VGPR v, lane half -> row
        int row = 16 * wave + m;
        if (ncol < NCLASS) out[row * NCLASS + ncol] = C[v] + bval;
    }
}

extern "C" void kernel_launch(void* const* d_in, const int* in_sizes, int n_in,
                              void* d_out, int out_size, void* d_ws, size_t ws_size,
                              hipStream_t stream) {
    const float* x = (const float*)d_in[0];     // [64, 196]
    const float* W = (const float*)d_in[1];     // [10, 14]
    const float* b = (const float*)d_in[2];     // [10]
    float* out  = (float*)d_out;                // [64, 10]
    float* qout = (float*)d_ws;                 // [64, 16] padded expectations

    // 2 state buffers (256KB) + params/reduction scratch: ~257KB of CDNA5's 320KB LDS
    const size_t shmem = (size_t)(4 * DIM + 200 + 16) * sizeof(float);
    hipFuncSetAttribute((const void*)qvc_sim,
                        hipFuncAttributeMaxDynamicSharedMemorySize, (int)shmem);

    qvc_sim<<<BATCH, SIM_THREADS, shmem, stream>>>(x, qout);
    qvc_linear<<<1, 128, 0, stream>>>(qout, W, b, out);
}
